// GNN_43422119362649
// MI455X (gfx1250) — matrix operand order repaired and checked by reading, not compiled, not run on hardware
//
#include <hip/hip_runtime.h>
#include <hip/hip_bf16.h>
#include <math.h>

typedef __attribute__((ext_vector_type(16))) _Float16 v16h;
typedef __attribute__((ext_vector_type(8)))  _Float16 v8h;
typedef __attribute__((ext_vector_type(8)))  float    v8f;

#define N_NODES 50000
#define N_EDGES 800000
#define HDIM 64

__device__ __forceinline__ float gnn_gelu(float x) {
    const float c0 = 0.7978845608028654f; // sqrt(2/pi)
    float x3 = x * x * x;
    return 0.5f * x * (1.0f + tanhf(c0 * (x + 0.044715f * x3)));
}

// ---------------------------------------------------------------------------
// Weight packing: reorder a KxN(=64) f32 row-major matrix into WMMA B-fragment
// order (f16): dst[((kc*4+nt)*32 + lane)*16 + t] = W[(kc*32 + (lane/16)*16 + t), nt*16 + lane%16]
// so at use time each lane reads 16 contiguous halves (two b128 loads).
// ---------------------------------------------------------------------------
__global__ void gnn_pack_w(const float* __restrict__ src, _Float16* __restrict__ dst, int total) {
    int idx = blockIdx.x * blockDim.x + threadIdx.x;
    if (idx >= total) return;
    int t    = idx & 15;
    int lane = (idx >> 4) & 31;
    int blk  = idx >> 9;
    int nt   = blk & 3;
    int kc   = blk >> 2;
    int K = kc * 32 + ((lane >> 4) << 4) + t;
    int n = nt * 16 + (lane & 15);
    dst[idx] = (_Float16)src[K * HDIM + n];
}

// ---------------------------------------------------------------------------
// Node embedding: x = nodes @ W_emb + b_emb   (N x 7) @ (7 x 64)
// ---------------------------------------------------------------------------
__global__ void gnn_embed(const float* __restrict__ nodes, const float* __restrict__ W,
                          const float* __restrict__ b, float* __restrict__ x, int N) {
    int idx = blockIdx.x * blockDim.x + threadIdx.x;
    if (idx >= N * HDIM) return;
    int n = idx >> 6, h = idx & 63;
    float acc = b[h];
#pragma unroll
    for (int d = 0; d < 7; ++d) acc += nodes[n * 7 + d] * W[d * HDIM + h];
    x[idx] = acc;
}

// ---------------------------------------------------------------------------
// Core WMMA tile: one wave computes a 16xK @ Kx64 -> 16x64 f32 tile.
// lds_in: f16 rows, stride KIN halves. wp: fragment-packed f16 weights.
// ---------------------------------------------------------------------------
template <int KIN>
__device__ __forceinline__ void gnn_mma_tile(const _Float16* lds_in, int mrow0,
                                             const _Float16* __restrict__ wp, v8f c[4]) {
    int lane = threadIdx.x & 31;
    int m  = lane & 15;
    int kb = (lane >> 4) << 4; // 0 or 16
#pragma unroll
    for (int kc = 0; kc < KIN / 32; ++kc) {
        const _Float16* arow = lds_in + (mrow0 + m) * KIN + kc * 32 + kb;
        v8h alo = *(const v8h*)(arow);
        v8h ahi = *(const v8h*)(arow + 8);
        v16h a;
#pragma unroll
        for (int t = 0; t < 8; ++t) { a[t] = alo[t]; a[t + 8] = ahi[t]; }
#pragma unroll
        for (int nt = 0; nt < 4; ++nt) {
            const _Float16* bp = wp + (size_t)(((kc * 4 + nt) * 32) + lane) * 16;
            v8h blo = *(const v8h*)(bp);
            v8h bhi = *(const v8h*)(bp + 8);
            v16h bv;
#pragma unroll
            for (int t = 0; t < 8; ++t) { bv[t] = blo[t]; bv[t + 8] = bhi[t]; }
            c[nt] = __builtin_amdgcn_wmma_f32_16x16x32_f16(
                false, a, false, bv, (short)0, c[nt], false, false);
        }
    }
}

__device__ __forceinline__ void gnn_zero_acc(v8f c[4]) {
    v8f z = {0.f, 0.f, 0.f, 0.f, 0.f, 0.f, 0.f, 0.f};
#pragma unroll
    for (int nt = 0; nt < 4; ++nt) c[nt] = z;
}

// bias + gelu, store D-fragment (lane holds M=r+8*(lane/16), N=(lane%16)+16*nt) to f16 LDS
__device__ __forceinline__ void gnn_store_act(const v8f c[4], const float* __restrict__ bias,
                                              int mrow0, _Float16* dst) {
    int lane = threadIdx.x & 31;
    int mo = (lane >> 4) << 3;
    int n0 = lane & 15;
#pragma unroll
    for (int nt = 0; nt < 4; ++nt) {
        int col = nt * 16 + n0;
        float bv = bias[col];
#pragma unroll
        for (int r = 0; r < 8; ++r) {
            float v = gnn_gelu(c[nt][r] + bv);
            dst[(mrow0 + r + mo) * HDIM + col] = (_Float16)v;
        }
    }
}

// ---------------------------------------------------------------------------
// Edge kernel: gather -> 3-layer MLP (WMMA) -> atomic segment-sum of pre-norm
// edges into `received` -> layer-norm -> store edges (f16, in place).
// KIN = 128 (step 0: [sent|recv]) or 192 (steps 1,2: [edges|sent|recv]).
// 128 threads = 4 waves, 64 edges per block.
// ---------------------------------------------------------------------------
template <int KIN>
__global__ __launch_bounds__(128) void gnn_edge_step(
    const float* __restrict__ x, _Float16* __restrict__ edges,
    const int* __restrict__ senders, const int* __restrict__ receivers,
    const _Float16* __restrict__ wp0, const _Float16* __restrict__ wp1,
    const _Float16* __restrict__ wp2,
    const float* __restrict__ eb0, const float* __restrict__ eb1, const float* __restrict__ eb2,
    const float* __restrict__ ln_s, const float* __restrict__ ln_b,
    float* __restrict__ received, int E) {
    constexpr int ROWS = 64;
    __shared__ __align__(16) _Float16 lds_in[ROWS * KIN];
    __shared__ __align__(16) _Float16 lds_h1[ROWS * HDIM];
    __shared__ __align__(16) _Float16 lds_h2[ROWS * HDIM];
    __shared__ __align__(16) float    lds_out[ROWS * HDIM];

    int tid = threadIdx.x;
    int ebase = blockIdx.x * ROWS;

    // ---- stage gathered inputs (f32 -> f16) ----
    for (int idx = tid; idx < ROWS * KIN; idx += blockDim.x) {
        int r = idx / KIN, k = idx - r * KIN;
        int e = ebase + r;
        float v = 0.f;
        if (e < E) {
            if (KIN == 192) {
                if (k < HDIM)            v = (float)edges[(size_t)e * HDIM + k];
                else if (k < 2 * HDIM)   v = x[(size_t)senders[e]   * HDIM + (k - HDIM)];
                else                     v = x[(size_t)receivers[e] * HDIM + (k - 2 * HDIM)];
            } else {
                if (k < HDIM)            v = x[(size_t)senders[e]   * HDIM + k];
                else                     v = x[(size_t)receivers[e] * HDIM + (k - HDIM)];
            }
        }
        lds_in[idx] = (_Float16)v;
    }
    __syncthreads();

    int lane = tid & 31;
    int mrow0 = (tid >> 5) * 16;

    v8f c[4];
    gnn_zero_acc(c);
    gnn_mma_tile<KIN>(lds_in, mrow0, wp0, c);
    gnn_store_act(c, eb0, mrow0, lds_h1);
    __syncthreads();

    gnn_zero_acc(c);
    gnn_mma_tile<HDIM>(lds_h1, mrow0, wp1, c);
    gnn_store_act(c, eb1, mrow0, lds_h2);
    __syncthreads();

    gnn_zero_acc(c);
    gnn_mma_tile<HDIM>(lds_h2, mrow0, wp2, c);

    // ---- final bias (no act): scatter pre-norm to received, stash f32 tile ----
    {
        int mo = (lane >> 4) << 3;
        int n0 = lane & 15;
        int rnode[8];
        int evalid[8];
#pragma unroll
        for (int r = 0; r < 8; ++r) {
            int e = ebase + mrow0 + r + mo;
            evalid[r] = (e < E);
            rnode[r] = evalid[r] ? receivers[e] : 0;
        }
#pragma unroll
        for (int nt = 0; nt < 4; ++nt) {
            int col = nt * 16 + n0;
            float bv = eb2[col];
#pragma unroll
            for (int r = 0; r < 8; ++r) {
                float v = c[nt][r] + bv;
                lds_out[(mrow0 + r + mo) * HDIM + col] = v;
                if (evalid[r]) atomicAdd(&received[(size_t)rnode[r] * HDIM + col], v);
            }
        }
    }
    __syncthreads();

    // ---- per-edge layer norm, write f16 edges (in place) ----
    if (tid < ROWS) {
        int e = ebase + tid;
        if (e < E) {
            const float* rowp = &lds_out[tid * HDIM];
            float mu = 0.f;
#pragma unroll
            for (int k = 0; k < HDIM; ++k) mu += rowp[k];
            mu *= (1.0f / HDIM);
            float s2 = 0.f;
#pragma unroll
            for (int k = 0; k < HDIM; ++k) { float d = rowp[k] - mu; s2 += d * d; }
            float inv = rsqrtf(s2 * (1.0f / HDIM) + 1e-6f);
#pragma unroll
            for (int k = 0; k < HDIM; ++k) {
                float v = (rowp[k] - mu) * inv * ln_s[k] + ln_b[k];
                edges[(size_t)e * HDIM + k] = (_Float16)v;
            }
        }
    }
}

// ---------------------------------------------------------------------------
// Node kernel: [x | received] -> 3-layer MLP (WMMA) -> layer-norm -> x (f32,
// in place). KIN fixed at 128. 64 nodes per 128-thread block.
// ---------------------------------------------------------------------------
__global__ __launch_bounds__(128) void gnn_node_step(
    float* __restrict__ x, const float* __restrict__ received,
    const _Float16* __restrict__ wp0, const _Float16* __restrict__ wp1,
    const _Float16* __restrict__ wp2,
    const float* __restrict__ nb0, const float* __restrict__ nb1, const float* __restrict__ nb2,
    const float* __restrict__ ln_s, const float* __restrict__ ln_b, int N) {
    constexpr int ROWS = 64;
    constexpr int KIN = 128;
    __shared__ __align__(16) _Float16 lds_in[ROWS * KIN];
    __shared__ __align__(16) _Float16 lds_h1[ROWS * HDIM];
    __shared__ __align__(16) _Float16 lds_h2[ROWS * HDIM];
    __shared__ __align__(16) float    lds_out[ROWS * HDIM];

    int tid = threadIdx.x;
    int nbase = blockIdx.x * ROWS;

    for (int idx = tid; idx < ROWS * KIN; idx += blockDim.x) {
        int r = idx >> 7, k = idx & 127;
        int n = nbase + r;
        float v = 0.f;
        if (n < N) v = (k < HDIM) ? x[(size_t)n * HDIM + k]
                                  : received[(size_t)n * HDIM + (k - HDIM)];
        lds_in[idx] = (_Float16)v;
    }
    __syncthreads();

    int mrow0 = (tid >> 5) * 16;
    v8f c[4];
    gnn_zero_acc(c);
    gnn_mma_tile<KIN>(lds_in, mrow0, wp0, c);
    gnn_store_act(c, nb0, mrow0, lds_h1);
    __syncthreads();

    gnn_zero_acc(c);
    gnn_mma_tile<HDIM>(lds_h1, mrow0, wp1, c);
    gnn_store_act(c, nb1, mrow0, lds_h2);
    __syncthreads();

    gnn_zero_acc(c);
    gnn_mma_tile<HDIM>(lds_h2, mrow0, wp2, c);

    // final bias (no act) into f32 tile
    {
        int lane = tid & 31;
        int mo = (lane >> 4) << 3;
        int n0 = lane & 15;
#pragma unroll
        for (int nt = 0; nt < 4; ++nt) {
            int col = nt * 16 + n0;
            float bv = nb2[col];
#pragma unroll
            for (int r = 0; r < 8; ++r)
                lds_out[(mrow0 + r + mo) * HDIM + col] = c[nt][r] + bv;
        }
    }
    __syncthreads();

    if (tid < ROWS) {
        int n = nbase + tid;
        if (n < N) {
            const float* rowp = &lds_out[tid * HDIM];
            float mu = 0.f;
#pragma unroll
            for (int k = 0; k < HDIM; ++k) mu += rowp[k];
            mu *= (1.0f / HDIM);
            float s2 = 0.f;
#pragma unroll
            for (int k = 0; k < HDIM; ++k) { float d = rowp[k] - mu; s2 += d * d; }
            float inv = rsqrtf(s2 * (1.0f / HDIM) + 1e-6f);
#pragma unroll
            for (int k = 0; k < HDIM; ++k)
                x[(size_t)n * HDIM + k] = (rowp[k] - mu) * inv * ln_s[k] + ln_b[k];
        }
    }
}

// ---------------------------------------------------------------------------
// Readout: column sums of x -> agg, then tiny MLP 64->256->128->128->1.
// ---------------------------------------------------------------------------
__global__ void gnn_colsum(const float* __restrict__ x, float* __restrict__ agg, int N) {
    __shared__ float part[HDIM];
    int tid = threadIdx.x; // 256
    if (tid < HDIM) part[tid] = 0.f;
    __syncthreads();
    int col = tid & 63;
    float acc = 0.f;
    int rowsPerBlk = blockDim.x / HDIM; // 4
    for (int r = blockIdx.x * rowsPerBlk + (tid >> 6); r < N; r += gridDim.x * rowsPerBlk)
        acc += x[(size_t)r * HDIM + col];
    atomicAdd(&part[col], acc);
    __syncthreads();
    if (tid < HDIM) atomicAdd(&agg[tid], part[tid]);
}

__global__ void gnn_readout(const float* __restrict__ agg,
                            const float* __restrict__ rW1, const float* __restrict__ rb1,
                            const float* __restrict__ rW2, const float* __restrict__ rb2,
                            const float* __restrict__ rW3, const float* __restrict__ rb3,
                            const float* __restrict__ rW4, const float* __restrict__ rb4,
                            float* __restrict__ out, float invN) {
    __shared__ float a[64], h1[256], h2[128], h3[128];
    int t = threadIdx.x; // 256
    if (t < 64) a[t] = agg[t] * invN;
    __syncthreads();
    {
        float s = rb1[t];
        for (int k = 0; k < 64; ++k) s += a[k] * rW1[k * 256 + t];
        h1[t] = gnn_gelu(s);
    }
    __syncthreads();
    if (t < 128) {
        float s = rb2[t];
        for (int k = 0; k < 256; ++k) s += h1[k] * rW2[k * 128 + t];
        h2[t] = gnn_gelu(s);
    }
    __syncthreads();
    if (t < 128) {
        float s = rb3[t];
        for (int k = 0; k < 128; ++k) s += h2[k] * rW3[k * 128 + t];
        h3[t] = gnn_gelu(s);
    }
    __syncthreads();
    if (t == 0) {
        float s = rb4[0];
        for (int k = 0; k < 128; ++k) s += h3[k] * rW4[k];
        out[0] = s;
    }
}

// ---------------------------------------------------------------------------
extern "C" void kernel_launch(void* const* d_in, const int* in_sizes, int n_in,
                              void* d_out, int out_size, void* d_ws, size_t ws_size,
                              hipStream_t stream) {
    const float* nodes     = (const float*)d_in[0];
    const int*   senders   = (const int*)d_in[1];
    const int*   receivers = (const int*)d_in[2];
    const float* W_emb     = (const float*)d_in[3];
    const float* b_emb     = (const float*)d_in[4];
    const float* eW0_first = (const float*)d_in[5];
    const float* eW0_rest  = (const float*)d_in[6];   // [2,192,64]
    const float* eW_hid    = (const float*)d_in[7];   // [3,2,64,64]
    const float* eb        = (const float*)d_in[8];   // [3,3,64]
    const float* nW0       = (const float*)d_in[9];   // [3,128,64]
    const float* nW_hid    = (const float*)d_in[10];  // [3,2,64,64]
    const float* nb        = (const float*)d_in[11];  // [3,3,64]
    const float* ln_ns     = (const float*)d_in[12];
    const float* ln_nb     = (const float*)d_in[13];
    const float* ln_es     = (const float*)d_in[14];
    const float* ln_eb     = (const float*)d_in[15];
    const float* rW1 = (const float*)d_in[16]; const float* rb1 = (const float*)d_in[17];
    const float* rW2 = (const float*)d_in[18]; const float* rb2 = (const float*)d_in[19];
    const float* rW3 = (const float*)d_in[20]; const float* rb3 = (const float*)d_in[21];
    const float* rW4 = (const float*)d_in[22]; const float* rb4 = (const float*)d_in[23];

    const int N = N_NODES, E = N_EDGES;

    // ---- workspace layout ----
    float*     x        = (float*)d_ws;                       // N*64 f32
    float*     received = x + (size_t)N * HDIM;               // N*64 f32
    _Float16*  edges    = (_Float16*)(received + (size_t)N * HDIM); // E*64 f16
    _Float16*  wpack    = edges + (size_t)E * HDIM;           // 106496 halves
    float*     agg      = (float*)(wpack + 106496);           // 64 f32

    // packed weight offsets (halves)
    const size_t WP_E0F   = 0;                 // eW0_first          (128x64)
    const size_t WP_E0R0  = 8192;              // eW0_rest[0]        (192x64)
    const size_t WP_E0R1  = 20480;             // eW0_rest[1]        (192x64)
    const size_t WP_EHID  = 32768;             // eW_hid[s][l]  6 x (64x64)
    const size_t WP_N0    = 57344;             // nW0[s]        3 x (128x64)
    const size_t WP_NHID  = 81920;             // nW_hid[s][l]  6 x (64x64)

    // ---- pack all MLP weights into WMMA B-fragment order (f16) ----
    gnn_pack_w<<<(128 * 64 + 255) / 256, 256, 0, stream>>>(eW0_first, wpack + WP_E0F, 128 * 64);
    gnn_pack_w<<<(192 * 64 + 255) / 256, 256, 0, stream>>>(eW0_rest,            wpack + WP_E0R0, 192 * 64);
    gnn_pack_w<<<(192 * 64 + 255) / 256, 256, 0, stream>>>(eW0_rest + 192 * 64, wpack + WP_E0R1, 192 * 64);
    for (int i = 0; i < 6; ++i) {
        gnn_pack_w<<<(64 * 64 + 255) / 256, 256, 0, stream>>>(
            eW_hid + (size_t)i * 64 * 64, wpack + WP_EHID + (size_t)i * 4096, 64 * 64);
        gnn_pack_w<<<(64 * 64 + 255) / 256, 256, 0, stream>>>(
            nW_hid + (size_t)i * 64 * 64, wpack + WP_NHID + (size_t)i * 4096, 64 * 64);
    }
    for (int s = 0; s < 3; ++s)
        gnn_pack_w<<<(128 * 64 + 255) / 256, 256, 0, stream>>>(
            nW0 + (size_t)s * 128 * 64, wpack + WP_N0 + (size_t)s * 8192, 128 * 64);

    // ---- embedding ----
    gnn_embed<<<(N * HDIM + 255) / 256, 256, 0, stream>>>(nodes, W_emb, b_emb, x, N);

    // ---- message passing steps ----
    int eBlocks = (E + 63) / 64;
    int nBlocks = (N + 63) / 64;
    for (int s = 0; s < 3; ++s) {
        hipMemsetAsync(received, 0, (size_t)N * HDIM * sizeof(float), stream);
        const _Float16* wph0 = wpack + WP_EHID + (size_t)(s * 2 + 0) * 4096;
        const _Float16* wph1 = wpack + WP_EHID + (size_t)(s * 2 + 1) * 4096;
        const float* eb0 = eb + (s * 3 + 0) * HDIM;
        const float* eb1 = eb + (s * 3 + 1) * HDIM;
        const float* eb2 = eb + (s * 3 + 2) * HDIM;
        if (s == 0) {
            gnn_edge_step<128><<<eBlocks, 128, 0, stream>>>(
                x, edges, senders, receivers, wpack + WP_E0F, wph0, wph1,
                eb0, eb1, eb2, ln_es + s * HDIM, ln_eb + s * HDIM, received, E);
        } else {
            const _Float16* wp0 = wpack + (s == 1 ? WP_E0R0 : WP_E0R1);
            gnn_edge_step<192><<<eBlocks, 128, 0, stream>>>(
                x, edges, senders, receivers, wp0, wph0, wph1,
                eb0, eb1, eb2, ln_es + s * HDIM, ln_eb + s * HDIM, received, E);
        }
        gnn_node_step<<<nBlocks, 128, 0, stream>>>(
            x, received,
            wpack + WP_N0 + (size_t)s * 8192,
            wpack + WP_NHID + (size_t)(s * 2 + 0) * 4096,
            wpack + WP_NHID + (size_t)(s * 2 + 1) * 4096,
            nb + (s * 3 + 0) * HDIM, nb + (s * 3 + 1) * HDIM, nb + (s * 3 + 2) * HDIM,
            ln_ns + s * HDIM, ln_nb + s * HDIM, N);
    }

    // ---- readout ----
    hipMemsetAsync(agg, 0, HDIM * sizeof(float), stream);
    gnn_colsum<<<128, 256, 0, stream>>>(x, agg, N);
    gnn_readout<<<1, 256, 0, stream>>>(agg, rW1, rb1, rW2, rb2, rW3, rb3, rW4, rb4,
                                       (float*)d_out, 1.0f / (float)N);
}
